// Attention_40767829574453
// MI455X (gfx1250) — matrix operand order repaired and checked by reading, not compiled, not run on hardware
//
#include <hip/hip_runtime.h>
#include <hip/hip_bf16.h>
#include <stdint.h>

// ---------------- problem constants ----------------
#define NB    16
#define TXD   256
#define TDEC  1024
#define EH    256      // ENC_HIDDEN
#define MEL   80
#define AH    128      // ATT_HIDDEN
#define VOC   100
#define SPKR  64
#define HC    256      // SPEC_ENC_H
#define KPAD  768      // im2col K = 3*256 (layer0 channel-padded to 256)
#define CSTR  256      // channel stride of time-major activation buffer
#define NEGV  (-1000000000.0f)

typedef __attribute__((ext_vector_type(16))) __bf16 v16bf;
typedef __attribute__((ext_vector_type(8)))  __bf16 v8bf;
typedef __attribute__((ext_vector_type(8)))  float  v8f;

// ---------------- WMMA helpers (CDNA5 16x16x32 bf16, f32 acc) ----------------
// A fragment (16x32, M x K), row-major source with leading dim lda (multiple of 16).
// Lane layout: lane<16 -> M=lane, K elems {0..15};
//              lane>=16 -> M=lane-16, K elems {16..31}.
__device__ __forceinline__ v16bf ld_a16(const __bf16* A, int lda, int row0, int k0) {
  int lane = threadIdx.x & 31;
  const __bf16* p = A + (size_t)(row0 + (lane & 15)) * lda + k0 + ((lane >> 4) & 1) * 16;
  v8bf lo = *(const v8bf*)p;
  v8bf hi = *(const v8bf*)(p + 8);
  v16bf r;
#pragma unroll
  for (int e = 0; e < 8; ++e) { r[e] = lo[e]; r[e + 8] = hi[e]; }
  return r;
}

// B fragment (32x16, K x N) where B is given as row-major [N][K] (i.e. we multiply by B^T).
// Lane layout: N = lane&15, K elems = (lane>>4)*16 + e  -> 16 contiguous halfwords per lane.
__device__ __forceinline__ v16bf ld_bT16(const __bf16* B, int ldb, int col0, int k0) {
  int lane = threadIdx.x & 31;
  const __bf16* p = B + (size_t)(col0 + (lane & 15)) * ldb + k0 + ((lane >> 4) & 1) * 16;
  return *(const v16bf*)p;
}

__device__ __forceinline__ v8f wmma_bf16(v16bf a, v16bf b, v8f c) {
  // (neg_a, A, neg_b, B, c_mod, C, reuse_a, reuse_b)
  return __builtin_amdgcn_wmma_f32_16x16x32_bf16(false, a, false, b, (short)0, c, false, false);
}

__device__ __forceinline__ float logsigf(float x) {
  return fminf(x, 0.f) - log1pf(expf(-fabsf(x)));
}
__device__ __forceinline__ float lae(float a, float b) {   // logaddexp
  float mx = fmaxf(a, b), mn = fminf(a, b);
  return mx + log1pf(expf(mn - mx));
}

// ---------------- init: zero attention + aux path, fill att_mask ----------------
__global__ void fill3_kernel(const int* __restrict__ tl, const int* __restrict__ sl,
                             float* __restrict__ att, float* __restrict__ amask,
                             float* __restrict__ paux) {
  size_t idx = (size_t)blockIdx.x * blockDim.x + threadIdx.x;
  if (idx >= (size_t)NB * TXD * TDEC) return;
  int n = (int)(idx >> 18);
  int rem = (int)(idx & 262143);
  int i = rem >> 10, j = rem & 1023;
  amask[idx] = (i < tl[n] && j < sl[n]) ? 1.f : 0.f;
  att[idx]  = 0.f;
  paux[idx] = 0.f;
}

// ---------------- pack conv weights: (cout,cin,3) f32 -> [256][768] bf16, q = k*256 + ci ----------------
// Channel dimension padded to 256 with zeros so every layer uses the same K=768 GEMM.
__global__ void pack_w_kernel(const float* __restrict__ w, int cin, __bf16* __restrict__ Wp) {
  int idx = blockIdx.x * blockDim.x + threadIdx.x;
  if (idx >= 256 * KPAD) return;
  int co = idx / KPAD, q = idx - co * KPAD;
  int k = q >> 8, ci = q & 255;
  float v = (ci < cin) ? w[((size_t)co * cin + ci) * 3 + k] : 0.f;
  Wp[idx] = (__bf16)v;
}

__global__ void pack_cast_kernel(const float* __restrict__ src, __bf16* __restrict__ dst, int count) {
  int idx = blockIdx.x * blockDim.x + threadIdx.x;
  if (idx < count) dst[idx] = (__bf16)src[idx];
}

__global__ void pack_pad_kernel(const float* __restrict__ src, __bf16* __restrict__ dst,
                                int srcRows, int dstRows, int cols) {
  int idx = blockIdx.x * blockDim.x + threadIdx.x;
  if (idx >= dstRows * cols) return;
  int r = idx / cols, c = idx - r * cols;
  dst[idx] = (__bf16)((r < srcRows) ? src[(size_t)r * cols + c] : 0.f);
}

// ---------------- per-layer conditioning bias: relu(mlp(spkr)) + relu(mlp(gst)) ----------------
__global__ void bias_mlp_kernel(const float* __restrict__ spkr, const float* __restrict__ gst,
                                const float* s1w, const float* s1b, const float* s2w, const float* s2b,
                                const float* g1w, const float* g1b, const float* g2w, const float* g2b,
                                float* __restrict__ biasOut) {
  int n = blockIdx.x, tid = threadIdx.x;
  __shared__ float h1[SPKR], h2[SPKR];
  if (tid < SPKR) {
    float a = s1b[tid];
    for (int j = 0; j < SPKR; ++j) a += spkr[n * SPKR + j] * s1w[tid * SPKR + j];
    h1[tid] = fmaxf(a, 0.f);
    float g = g1b[tid];
    for (int j = 0; j < EH; ++j) g += gst[n * EH + j] * g1w[tid * EH + j];
    h2[tid] = fmaxf(g, 0.f);
  }
  __syncthreads();
  float a = s2b[tid], g = g2b[tid];
  for (int j = 0; j < SPKR; ++j) { a += h1[j] * s2w[tid * SPKR + j]; g += h2[j] * g2w[tid * SPKR + j]; }
  biasOut[n * HC + tid] = fmaxf(a, 0.f) + fmaxf(g, 0.f);
}

// ---------------- initial activation pack: spec (n,t,80) -> masked bf16 xT [n][T+2][256] ----------------
__global__ void pack_spec_kernel(const float* __restrict__ spec, const int* __restrict__ sl,
                                 __bf16* __restrict__ xT) {
  size_t idx = (size_t)blockIdx.x * blockDim.x + threadIdx.x;
  if (idx >= (size_t)NB * TDEC * CSTR) return;
  int c = (int)(idx & 255);
  int t = (int)((idx >> 8) & 1023);
  int n = (int)(idx >> 18);
  float v = 0.f;
  if (c < MEL && t < sl[n]) v = spec[((size_t)n * TDEC + t) * MEL + c];
  __bf16* xb = xT + (size_t)n * (TDEC + 2) * CSTR;
  xb[(size_t)(t + 1) * CSTR + c] = (__bf16)v;
  if (t == 0)        xb[c]                              = (__bf16)0.f;
  if (t == TDEC - 1) xb[(size_t)(TDEC + 1) * CSTR + c]  = (__bf16)0.f;
}

// ---------------- conv-as-GEMM, fully unrolled K=768; 1 M-tile x 4 N-tiles per wave ----------------
// Y[n][co][t] = sum_{k,ci} W[co][k*256+ci] * xT[n][t+k][ci]   (xT has +1 pad row)
__global__ void conv_gemm_kernel(const __bf16* __restrict__ Wp, const __bf16* __restrict__ xT,
                                 float* __restrict__ Y) {
  int wave = blockIdx.x * (blockDim.x >> 5) + (threadIdx.x >> 5);
  if (wave >= NB * 16 * 16) return;
  int n = wave >> 8;
  int r = wave & 255;
  int ct = r >> 4, tg = r & 15;          // ct: co tile, tg: group of 4 t-tiles
  int lane = threadIdx.x & 31;
  int g = (lane >> 4) & 1;
  int tbase = tg * 64 + (lane & 15);
  const __bf16* xb = xT + (size_t)n * (TDEC + 2) * CSTR;
  v8f a0 = {}, a1 = {}, a2 = {}, a3 = {};
#pragma unroll
  for (int s = 0; s < 24; ++s) {
    int k0 = s * 32;
    v16bf a = ld_a16(Wp, KPAD, ct * 16, k0);
    int kk  = (k0 + g * 16) >> 8;        // conv tap (0..2), uniform within the fragment half
    int cib = (k0 + g * 16) & 255;       // channel base, multiple of 16 -> 32B aligned
    const __bf16* bp = xb + (size_t)(tbase + kk) * CSTR + cib;
    v16bf b0 = *(const v16bf*)bp;
    v16bf b1 = *(const v16bf*)(bp + 16 * CSTR);
    v16bf b2 = *(const v16bf*)(bp + 32 * CSTR);
    v16bf b3 = *(const v16bf*)(bp + 48 * CSTR);
    a0 = wmma_bf16(a, b0, a0);
    a1 = wmma_bf16(a, b1, a1);
    a2 = wmma_bf16(a, b2, a2);
    a3 = wmma_bf16(a, b3, a3);
  }
  float* Yn = Y + (size_t)n * HC * TDEC;
#pragma unroll
  for (int rr = 0; rr < 8; ++rr) {
    int co = ct * 16 + rr + g * 8;
    float* yr = Yn + (size_t)co * TDEC + tbase;
    yr[0]  = a0[rr];
    yr[16] = a1[rr];
    yr[32] = a2[rr];
    yr[48] = a3[rr];
  }
}

// ---------------- GroupNorm stats over time, per (n,c): one wave per row ----------------
__global__ void gn_stats_kernel(const float* __restrict__ Y, float* __restrict__ mean,
                                float* __restrict__ rstd) {
  int wave = blockIdx.x * (blockDim.x >> 5) + (threadIdx.x >> 5);
  if (wave >= NB * HC) return;
  const float* row = Y + (size_t)wave * TDEC;
  int lane = threadIdx.x & 31;
  float s = 0.f, s2 = 0.f;
  for (int t = lane; t < TDEC; t += 32) { float v = row[t]; s += v; s2 += v * v; }
  for (int off = 16; off; off >>= 1) { s += __shfl_down(s, off, 32); s2 += __shfl_down(s2, off, 32); }
  if (lane == 0) {
    float mu = s / (float)TDEC;
    float var = s2 / (float)TDEC - mu * mu;
    mean[wave] = mu;
    rstd[wave] = rsqrtf(var + 1e-5f);
  }
}

// ---------------- GN + relu + cond-bias (+residual); write next-layer xT or final transposed out ----------------
// Note: conv bias cancels inside GroupNorm (mean subtraction) -> omitted.
__global__ void gn_post_kernel(const float* __restrict__ Y, const float* __restrict__ mean,
                               const float* __restrict__ rstd,
                               const float* __restrict__ gnw, const float* __restrict__ gnb,
                               const float* __restrict__ bias, const float* __restrict__ xres,
                               const int* __restrict__ sl,
                               float* __restrict__ xmask, __bf16* __restrict__ xT,
                               __bf16* __restrict__ outT, int is_final) {
  size_t idx = (size_t)blockIdx.x * blockDim.x + threadIdx.x;
  if (idx >= (size_t)NB * HC * TDEC) return;
  int t = (int)(idx & 1023);
  int nc = (int)(idx >> 10);
  int c = nc & 255, n = nc >> 8;
  float y = (Y[idx] - mean[nc]) * rstd[nc] * gnw[c] + gnb[c];
  y = fmaxf(y, 0.f);
  float val = y + bias[n * HC + c] + (xres ? xres[idx] : 0.f);
  if (is_final) {
    outT[((size_t)n * TDEC + t) * HC + c] = (__bf16)val;   // (n,t,c) for downstream GEMMs
  } else {
    float m = (t < sl[n]) ? 1.f : 0.f;
    float vm = val * m;
    xmask[idx] = vm;
    __bf16* xb = xT + (size_t)n * (TDEC + 2) * CSTR;
    xb[(size_t)(t + 1) * CSTR + c] = (__bf16)vm;
    if (t == 0)        xb[c]                             = (__bf16)0.f;
    if (t == TDEC - 1) xb[(size_t)(TDEC + 1) * CSTR + c] = (__bf16)0.f;
  }
}

// ---------------- generic C = A * B^T (bf16 in, f32 out), 1 tile per wave ----------------
__global__ void gemm_abT_kernel(const __bf16* __restrict__ A, int lda, long long sA,
                                const __bf16* __restrict__ B, int ldb, long long sB,
                                float* __restrict__ C, int ldc, long long sC,
                                int Mtiles, int Ntiles, int nBatch, int K, int Nstore,
                                const float* __restrict__ bias,
                                const float* __restrict__ sw, const float* __restrict__ sb) {
  int wave = blockIdx.x * (blockDim.x >> 5) + (threadIdx.x >> 5);
  int tpb = Mtiles * Ntiles;
  if (wave >= nBatch * tpb) return;
  int b = wave / tpb;
  int r = wave - b * tpb;
  int mt = r / Ntiles, nt = r - (r / Ntiles) * Ntiles;
  const __bf16* Ab = A + (size_t)b * sA;
  const __bf16* Bb = B + (size_t)b * sB;
  float* Cb = C + (size_t)b * sC;
  v8f acc = {};
#pragma unroll 4
  for (int k0 = 0; k0 < K; k0 += 32) {
    if (k0 + 32 < K) __builtin_prefetch(Ab + (size_t)(mt * 16) * lda + k0 + 32, 0, 0);
    v16bf a = ld_a16(Ab, lda, mt * 16, k0);
    v16bf bb = ld_bT16(Bb, ldb, nt * 16, k0);
    acc = wmma_bf16(a, bb, acc);
  }
  float scale = 1.f, shift = 0.f;
  if (sw) { scale = 10.f * expf(sw[0]); shift = sb[0]; }   // similarity epilogue
  int lane = threadIdx.x & 31;
  int n = nt * 16 + (lane & 15);
  if (n < Nstore) {
    float bv = bias ? bias[n] : 0.f;
#pragma unroll
    for (int rr = 0; rr < 8; ++rr) {
      int m = mt * 16 + rr + ((lane >> 4) & 1) * 8;
      Cb[(size_t)m * ldc + n] = acc[rr] * scale + shift + bv;
    }
  }
}

// ---------------- L2-normalize rows, emit bf16 (key) ----------------
__global__ void norm_pack_kernel(const float* __restrict__ X, __bf16* __restrict__ out,
                                 int rows, int cols) {
  int wave = blockIdx.x * (blockDim.x >> 5) + (threadIdx.x >> 5);
  if (wave >= rows) return;
  int lane = threadIdx.x & 31;
  const float* row = X + (size_t)wave * cols;
  float ss = 0.f;
  for (int c = lane; c < cols; c += 32) { float v = row[c]; ss += v * v; }
  for (int off = 16; off; off >>= 1) ss += __shfl_down(ss, off, 32);
  ss = __shfl(ss, 0, 32);
  float denom = fmaxf(sqrtf(ss), 1e-8f);
  for (int c = lane; c < cols; c += 32) out[(size_t)wave * cols + c] = (__bf16)(row[c] / denom);
}

// ---------------- query: append energy = mean(spec,mel) as col 127, normalize, pack bf16 ----------------
__global__ void query_fin_kernel(const float* __restrict__ qf, const float* __restrict__ spec,
                                 __bf16* __restrict__ qb) {
  int wave = blockIdx.x * (blockDim.x >> 5) + (threadIdx.x >> 5);
  if (wave >= NB * TDEC) return;
  int lane = threadIdx.x & 31;
  int n = wave >> 10, t = wave & 1023;
  float e = 0.f;
  for (int m = lane; m < MEL; m += 32) e += spec[((size_t)n * TDEC + t) * MEL + m];
  for (int off = 16; off; off >>= 1) e += __shfl_down(e, off, 32);
  e = __shfl(e, 0, 32) / (float)MEL;
  const float* row = qf + (size_t)wave * AH;
  float ss = 0.f;
  for (int c = lane; c < AH; c += 32) { float v = (c == AH - 1) ? e : row[c]; ss += v * v; }
  for (int off = 16; off; off >>= 1) ss += __shfl_down(ss, off, 32);
  ss = __shfl(ss, 0, 32);
  float denom = fmaxf(sqrtf(ss), 1e-8f);
  for (int c = lane; c < AH; c += 32) {
    float v = (c == AH - 1) ? e : row[c];
    qb[(size_t)wave * AH + c] = (__bf16)(v / denom);
  }
}

// ---------------- CTC softmax / log-softmax per (n,t) over 100 classes (stride 112) ----------------
__global__ void ctc_soft_kernel(const float* __restrict__ logit, float* __restrict__ prob,
                                float* __restrict__ lsm) {
  int wave = blockIdx.x * (blockDim.x >> 5) + (threadIdx.x >> 5);
  if (wave >= NB * TDEC) return;
  int lane = threadIdx.x & 31;
  const float* row = logit + (size_t)wave * 112;
  float mx = -1e30f;
  for (int v = lane; v < VOC; v += 32) mx = fmaxf(mx, row[v]);
  for (int off = 16; off; off >>= 1) mx = fmaxf(mx, __shfl_down(mx, off, 32));
  mx = __shfl(mx, 0, 32);
  float s = 0.f;
  for (int v = lane; v < VOC; v += 32) s += expf(row[v] - mx);
  for (int off = 16; off; off >>= 1) s += __shfl_down(s, off, 32);
  s = __shfl(s, 0, 32);
  float ls = logf(s);
  for (int v = lane; v < VOC; v += 32) {
    float z = row[v] - mx;
    prob[(size_t)wave * 112 + v] = expf(z) / s;
    lsm[(size_t)wave * 112 + v]  = z - ls;
  }
}

// ---------------- MAS forward DP: v[n][j][i]; value = logsig(sim)*mask (min-term cancels) ----------------
__global__ void mas_fwd_kernel(const float* __restrict__ sim, const float* __restrict__ cq,
                               const int* __restrict__ tok, const int* __restrict__ tl,
                               const int* __restrict__ sl, float* __restrict__ v) {
  int n = blockIdx.x, i = threadIdx.x;   // blockDim == 256 == TXD
  __shared__ float carry[TXD];
  carry[i] = NEGV;
  __syncthreads();
  int tx = tl[n], ty = sl[n];
  int tk = tok ? tok[n * TXD + i] : 0;
  for (int j = 0; j < TDEC; ++j) {
    float val = 0.f;
    if (i < tx && j < ty) {
      float s = sim ? sim[((size_t)n * TXD + i) * TDEC + j]
                    : cq[((size_t)n * TDEC + j) * 112 + tk];
      val = logsigf(s);
    }
    float vc = (i == j) ? NEGV : carry[i];
    float vp = (i == 0) ? ((j == 0) ? 0.f : NEGV) : carry[i - 1];
    float nv = val + fmaxf(vp, vc);
    __syncthreads();
    carry[i] = nv;
    v[((size_t)n << 18) + ((size_t)j << 8) + i] = nv;
    __syncthreads();
  }
}

// ---------------- MAS backward: scalar walk per batch, writes 1s into zeroed path ----------------
__global__ void mas_bwd_kernel(const float* __restrict__ v, const int* __restrict__ tl,
                               const int* __restrict__ sl, float* __restrict__ path) {
  int n = blockIdx.x * blockDim.x + threadIdx.x;
  if (n >= NB) return;
  int index = tl[n] - 1, ty = sl[n];
  for (int j = TDEC - 1; j >= 0; --j) {
    bool write = j < ty;
    if (write) path[((size_t)n * TXD + index) * TDEC + j] = 1.f;
    const float* vcol = v + ((size_t)n << 18) + ((size_t)((j > 0) ? j - 1 : 0) << 8);
    float vh = vcol[index];
    float vu = vcol[(index > 0) ? index - 1 : 0];
    bool move = (index != 0) && ((index == j) || (vu > vh));
    if (write && move) --index;
  }
}

// ---------------- BCE-style alignment losses, per batch ----------------
__global__ void loss_reduce_kernel(const float* __restrict__ sim, const float* __restrict__ att,
                                   const float* __restrict__ paux, const int* __restrict__ tl,
                                   const int* __restrict__ sl, float* __restrict__ icl,
                                   float* __restrict__ auxl) {
  int n = blockIdx.x, tid = threadIdx.x;
  int tx = tl[n], ty = sl[n];
  const size_t b = (size_t)n << 18;
  float s1 = 0.f, s2 = 0.f;
  for (int e = tid; e < TXD * TDEC; e += 256) {
    int i = e >> 10, j = e & 1023;
    float s = sim[b + e];
    float ls = logsigf(s), nls = logsigf(-s);
    float mask = (i < tx && j < ty) ? 1.f : 0.f;
    float mm = att[b + e];
    s1 += mm * ls + (1.f - mm) * mask * nls;
    float p0 = (i > 0)        ? paux[b + (size_t)((i - 1) << 10) + j] : 0.f;
    float p1 = paux[b + e];
    float p2 = (i < TXD - 1)  ? paux[b + (size_t)((i + 1) << 10) + j] : 0.f;
    float aa = (p0 + p1 + p2) * mask;
    s2 += aa * ls + (1.f - aa) * mask * nls;
  }
  __shared__ float r1[256], r2[256];
  r1[tid] = s1; r2[tid] = s2;
  __syncthreads();
  for (int off = 128; off; off >>= 1) {
    if (tid < off) { r1[tid] += r1[tid + off]; r2[tid] += r2[tid + off]; }
    __syncthreads();
  }
  if (tid == 0) {
    float denom = (float)tx * (float)ty;
    icl[n]  = -r1[0] / denom;
    auxl[n] = -r2[0] / denom * 0.5f;
  }
}

// ---------------- CTC forward algorithm, one block per batch, L = 513 states in LDS ----------------
__global__ void ctc_loss_kernel(const float* __restrict__ lsm, const int* __restrict__ tok,
                                const int* __restrict__ ilen, const int* __restrict__ tlen,
                                float* __restrict__ out) {
  int n = blockIdx.x, tid = threadIdx.x;   // blockDim == 512
  const int L = 2 * TXD + 1;
  __shared__ float A0[L], A1[L];
  __shared__ int ext[L];
  for (int l = tid; l < L; l += 512) ext[l] = (l & 1) ? tok[n * TXD + (l >> 1)] : 0;
  __syncthreads();
  const float* lp = lsm + (size_t)n * TDEC * 112;
  for (int l = tid; l < L; l += 512) A0[l] = (l < 2) ? lp[ext[l]] : NEGV;
  __syncthreads();
  float* cur = A0; float* nxt = A1;
  int il = ilen[n];
  for (int t = 1; t < TDEC; ++t) {
    bool active = t < il;
    for (int l = tid; l < L; l += 512) {
      float a = cur[l], o;
      if (active) {
        float bb = (l >= 1) ? cur[l - 1] : NEGV;
        bool skip = (l >= 2) && (ext[l] != 0) && (ext[l] != ext[l - 2]);
        float c = skip ? cur[l - 2] : NEGV;
        o = lae(lae(a, bb), c) + lp[(size_t)t * 112 + ext[l]];
      } else o = a;
      nxt[l] = o;
    }
    __syncthreads();
    float* tmp = cur; cur = nxt; nxt = tmp;
  }
  if (tid == 0) {
    int tl2 = tlen[n];
    float loss = -lae(cur[2 * tl2], cur[2 * tl2 - 1]);
    out[n] = loss / (float)tl2;
  }
}

__global__ void finalize_kernel(const float* __restrict__ icl, const float* __restrict__ auxl,
                                const float* __restrict__ ctcn, float* __restrict__ att_loss,
                                float* __restrict__ nll) {
  float a = 0.f, b = 0.f, c = 0.f;
  for (int n = 0; n < NB; ++n) { a += icl[n]; b += auxl[n]; c += ctcn[n]; }
  a /= NB; b /= NB; c /= NB;
  *nll = a;
  *att_loss = a + b + c;
}

// ================================================================
extern "C" void kernel_launch(void* const* d_in, const int* in_sizes, int n_in,
                              void* d_out, int out_size, void* d_ws, size_t ws_size,
                              hipStream_t stream) {
  (void)in_sizes; (void)n_in; (void)out_size; (void)ws_size;
  const float* text = (const float*)d_in[0];
  const float* spec = (const float*)d_in[1];
  const float* spkr = (const float*)d_in[2];
  const float* gst  = (const float*)d_in[3];
  const int* tlen   = (const int*)d_in[4];
  const int* slen   = (const int*)d_in[5];
  const int* tok    = (const int*)d_in[6];
  auto LP = [&](int s, int l, int f) { return (const float*)d_in[7 + (s * 6 + l) * 12 + f]; };
  const float* enc_w = (const float*)d_in[151];
  const float* enc_b = (const float*)d_in[152];
  const float* qp_w  = (const float*)d_in[153];
  const float* qp_b  = (const float*)d_in[154];
  const float* ctc_w = (const float*)d_in[155];
  const float* ctc_b = (const float*)d_in[156];
  const float* sim_w = (const float*)d_in[157];
  const float* sim_b = (const float*)d_in[158];

  // ---- workspace carve-up (256B aligned) ----
  char* base = (char*)d_ws;
  size_t off = 0;
  auto alloc = [&](size_t bytes) -> void* {
    void* p = base + off;
    off = (off + bytes + 255) & ~(size_t)255;
    return p;
  };
  __bf16* Wpack  = (__bf16*)alloc((size_t)12 * 256 * KPAD * 2);
  float*  bias12 = (float*) alloc((size_t)12 * NB * HC * 4);
  __bf16* xT     = (__bf16*)alloc((size_t)NB * (TDEC + 2) * CSTR * 2);
  float*  xmask  = (float*) alloc((size_t)NB * HC * TDEC * 4);
  float*  yconv  = (float*) alloc((size_t)NB * HC * TDEC * 4);
  float*  gmean  = (float*) alloc((size_t)NB * HC * 4);
  float*  grstd  = (float*) alloc((size_t)NB * HC * 4);
  __bf16* omb    = (__bf16*)alloc((size_t)NB * TDEC * HC * 2);
  __bf16* oab    = (__bf16*)alloc((size_t)NB * TDEC * HC * 2);
  __bf16* tb     = (__bf16*)alloc((size_t)NB * TXD * EH * 2);
  __bf16* Wb_enc = (__bf16*)alloc((size_t)AH * EH * 2);
  __bf16* Wb_q   = (__bf16*)alloc((size_t)128 * HC * 2);
  __bf16* Wb_ctc = (__bf16*)alloc((size_t)112 * HC * 2);
  float*  keyf   = (float*) alloc((size_t)NB * TXD * AH * 4);
  __bf16* keyb   = (__bf16*)alloc((size_t)NB * TXD * AH * 2);
  float*  qf     = (float*) alloc((size_t)NB * TDEC * AH * 4);
  __bf16* qb     = (__bf16*)alloc((size_t)NB * TDEC * AH * 2);
  float*  simb   = (float*) alloc((size_t)NB * TXD * TDEC * 4);
  float*  clogit = (float*) alloc((size_t)NB * TDEC * 112 * 4);
  float*  cq     = (float*) alloc((size_t)NB * TDEC * 112 * 4);
  float*  clsm   = (float*) alloc((size_t)NB * TDEC * 112 * 4);
  float*  vdp    = (float*) alloc((size_t)NB * TXD * TDEC * 4);
  float*  paux   = (float*) alloc((size_t)NB * TXD * TDEC * 4);
  float*  icl    = (float*) alloc(64);
  float*  auxl   = (float*) alloc(64);
  float*  ctcn   = (float*) alloc(64);

  float* attn     = (float*)d_out;
  float* att_loss = attn + (size_t)NB * TXD * TDEC;
  float* amask    = att_loss + 1;
  float* nll      = amask + (size_t)NB * TXD * TDEC;

  auto gw = [](long long waves) { return (int)((waves + 7) / 8); };  // 8 waves / 256-thread block

  // 1) init outputs + aux path
  fill3_kernel<<<(NB * TXD * TDEC) / 256, 256, 0, stream>>>(tlen, slen, attn, amask, paux);

  // 2) packs
  for (int s = 0; s < 2; ++s)
    for (int l = 0; l < 6; ++l)
      pack_w_kernel<<<(256 * KPAD) / 256, 256, 0, stream>>>(
          LP(s, l, 0), (l == 0) ? MEL : HC, Wpack + (size_t)(s * 6 + l) * 256 * KPAD);
  pack_cast_kernel<<<(NB * TXD * EH) / 256, 256, 0, stream>>>(text, tb, NB * TXD * EH);
  pack_cast_kernel<<<(AH * EH) / 256, 256, 0, stream>>>(enc_w, Wb_enc, AH * EH);
  pack_pad_kernel<<<(128 * HC) / 256, 256, 0, stream>>>(qp_w, Wb_q, AH - 1, 128, HC);
  pack_pad_kernel<<<(112 * HC) / 256, 256, 0, stream>>>(ctc_w, Wb_ctc, VOC, 112, HC);

  // 3) conditioning biases (12 layer instances)
  for (int s = 0; s < 2; ++s)
    for (int l = 0; l < 6; ++l)
      bias_mlp_kernel<<<NB, 256, 0, stream>>>(
          spkr, gst, LP(s, l, 4), LP(s, l, 5), LP(s, l, 6), LP(s, l, 7),
          LP(s, l, 8), LP(s, l, 9), LP(s, l, 10), LP(s, l, 11),
          bias12 + (size_t)(s * 6 + l) * NB * HC);

  // 4) both encoder stacks (main -> omb, aux -> oab)
  for (int s = 0; s < 2; ++s) {
    pack_spec_kernel<<<(NB * TDEC * CSTR) / 256, 256, 0, stream>>>(spec, slen, xT);
    for (int l = 0; l < 6; ++l) {
      conv_gemm_kernel<<<gw((long long)NB * 16 * 16), 256, 0, stream>>>(
          Wpack + (size_t)(s * 6 + l) * 256 * KPAD, xT, yconv);
      gn_stats_kernel<<<gw(NB * HC), 256, 0, stream>>>(yconv, gmean, grstd);
      int is_final = (l == 5);
      const float* xres = (l % 2 == 1) ? xmask : nullptr;
      __bf16* outT = is_final ? ((s == 0) ? omb : oab) : nullptr;
      gn_post_kernel<<<(NB * HC * TDEC) / 256, 256, 0, stream>>>(
          yconv, gmean, grstd, LP(s, l, 2), LP(s, l, 3),
          bias12 + (size_t)(s * 6 + l) * NB * HC, xres, slen, xmask, xT, outT, is_final);
    }
  }

  // 5) key = normalize(text @ enc_proj^T + b)
  gemm_abT_kernel<<<gw((long long)NB * 16 * 8), 256, 0, stream>>>(
      tb, EH, (long long)TXD * EH, Wb_enc, EH, 0, keyf, AH, (long long)TXD * AH,
      16, 8, NB, EH, AH, enc_b, nullptr, nullptr);
  norm_pack_kernel<<<gw(NB * TXD), 256, 0, stream>>>(keyf, keyb, NB * TXD, AH);

  // 6) query = normalize([out_main @ qproj^T + b, energy])
  gemm_abT_kernel<<<gw((long long)NB * 64 * 8), 256, 0, stream>>>(
      omb, HC, (long long)TDEC * HC, Wb_q, HC, 0, qf, AH, (long long)TDEC * AH,
      64, 8, NB, HC, AH - 1, qp_b, nullptr, nullptr);
  query_fin_kernel<<<gw(NB * TDEC), 256, 0, stream>>>(qf, spec, qb);

  // 7) similarity = 10*exp(sim_w) * key @ query^T + sim_b
  gemm_abT_kernel<<<gw((long long)NB * 16 * 64), 256, 0, stream>>>(
      keyb, AH, (long long)TXD * AH, qb, AH, (long long)TDEC * AH, simb, TDEC,
      (long long)TXD * TDEC, 16, 64, NB, AH, TDEC, nullptr, sim_w, sim_b);

  // 8) ctc logits -> softmax / log-softmax
  gemm_abT_kernel<<<gw((long long)NB * 64 * 7), 256, 0, stream>>>(
      oab, HC, (long long)TDEC * HC, Wb_ctc, HC, 0, clogit, 112, (long long)TDEC * 112,
      64, 7, NB, HC, VOC, ctc_b, nullptr, nullptr);
  ctc_soft_kernel<<<gw(NB * TDEC), 256, 0, stream>>>(clogit, cq, clsm);

  // 9) MAS on similarity -> attention; MAS on ctc alignment -> aux path
  mas_fwd_kernel<<<NB, TXD, 0, stream>>>(simb, nullptr, nullptr, tlen, slen, vdp);
  mas_bwd_kernel<<<1, 32, 0, stream>>>(vdp, tlen, slen, attn);
  mas_fwd_kernel<<<NB, TXD, 0, stream>>>(nullptr, cq, tok, tlen, slen, vdp);
  mas_bwd_kernel<<<1, 32, 0, stream>>>(vdp, tlen, slen, paux);

  // 10) losses
  loss_reduce_kernel<<<NB, 256, 0, stream>>>(simb, attn, paux, tlen, slen, icl, auxl);
  ctc_loss_kernel<<<NB, 512, 0, stream>>>(clsm, tok, slen, tlen, ctcn);
  finalize_kernel<<<1, 1, 0, stream>>>(icl, auxl, ctcn, att_loss, nll);
}
